// ParserLSTM_17652315586714
// MI455X (gfx1250) — compile-verified, matrix operand order checked by reading, not run on hardware
//
#include <hip/hip_runtime.h>

// Problem constants (from reference)
#define TT   256          // sequence length
#define BB   64           // batch
#define HH   512          // hidden
#define GATES 2048        // 4*H
#define INK  1024         // K of input GEMM (= IN = 2H)
#define NPROJ 4096        // 2 dirs * 4H gate columns
#define NWG_PER_DIR 16
#define HSLICE 32         // HH / NWG_PER_DIR

typedef __attribute__((ext_vector_type(16))) _Float16 v16h;
typedef __attribute__((ext_vector_type(8)))  _Float16 v8h;
typedef __attribute__((ext_vector_type(8)))  float    v8f;

// Load a 16x32 WMMA A/B fragment row (K-major). CDNA5 16-bit layout:
// lanes 0-15  hold K = {0..7} and {16..23}
// lanes 16-31 hold K = {8..15} and {24..31}
__device__ __forceinline__ v16h load_frag_f16(const _Float16* p, int hgrp) {
  v8h lo = *(const v8h*)(p + hgrp * 8);
  v8h hi = *(const v8h*)(p + 16 + hgrp * 8);
  v16h r;
#pragma unroll
  for (int i = 0; i < 8; ++i) { r[i] = lo[i]; r[8 + i] = hi[i]; }
  return r;
}

__device__ __forceinline__ float sigmoidf_(float x) {
  return 1.0f / (1.0f + __expf(-x));
}

// ---------------- utility kernels ----------------
__global__ void cvt_f32_f16(const float* __restrict__ src,
                            _Float16* __restrict__ dst, int n) {
  for (int i = blockIdx.x * blockDim.x + threadIdx.x; i < n;
       i += gridDim.x * blockDim.x)
    dst[i] = (_Float16)src[i];
}

__global__ void zero_u32(unsigned* __restrict__ p, int n) {
  for (int i = blockIdx.x * blockDim.x + threadIdx.x; i < n;
       i += gridDim.x * blockDim.x)
    p[i] = 0u;
}

// ---------------- input projection GEMM ----------------
// C[16384, 4096] (f16) = A[16384, 1024] (f16) * W[4096, 1024]^T (f16)
// Block tile 128x128, 8 waves (2x4), wave tile 64x32, K-step 32.
// Double-buffered LDS staged via CDNA5 async LDS-DMA (ASYNCcnt):
// tile k+1's global_load_async_to_lds_b128 overlaps tile k's WMMAs.
__global__ __launch_bounds__(256) void xproj_gemm_f16(
    const _Float16* __restrict__ A, const _Float16* __restrict__ W,
    _Float16* __restrict__ C) {
  __shared__ _Float16 As[2][128][32];
  __shared__ _Float16 Ws[2][128][32];

  const int tid  = threadIdx.x;
  const int wave = tid >> 5, lane = tid & 31;
  const int lr = lane & 15, hgrp = lane >> 4;
  const int wm = wave >> 2, wn = wave & 3;
  const int mBase = blockIdx.x * 128;
  const int nBase = blockIdx.y * 128;

  // Issue async DMA of one 128x32 A tile + 128x32 W tile into LDS buffer buf.
  // Each of 256 threads moves 2x16B of A and 2x16B of W.
  auto stage = [&](int kk, int buf) {
#pragma unroll
    for (int c = 0; c < 2; ++c) {
      int chunk = tid + c * 256;                 // 0..511
      int row   = chunk >> 2;
      int cx    = (chunk & 3) * 8;
      unsigned lA = (unsigned)(size_t)&As[buf][row][cx];  // LDS byte offset
      unsigned lW = (unsigned)(size_t)&Ws[buf][row][cx];
      const _Float16* gA = A + (size_t)(mBase + row) * INK + kk + cx;
      const _Float16* gW = W + (size_t)(nBase + row) * INK + kk + cx;
      asm volatile("global_load_async_to_lds_b128 %0, %1, off"
                   :: "v"(lA), "v"(gA) : "memory");
      asm volatile("global_load_async_to_lds_b128 %0, %1, off"
                   :: "v"(lW), "v"(gW) : "memory");
    }
  };

  const v8f vzero = {0.f,0.f,0.f,0.f,0.f,0.f,0.f,0.f};
  v8f acc[4][2];
#pragma unroll
  for (int i = 0; i < 4; ++i)
#pragma unroll
    for (int j = 0; j < 2; ++j) acc[i][j] = vzero;

  stage(0, 0);
  asm volatile("s_wait_asynccnt 0x0" ::: "memory");
  __syncthreads();

  const int NKT = INK / 32;
  for (int kt = 0; kt < NKT; ++kt) {
    const int cur = kt & 1;
    if (kt + 1 < NKT) stage((kt + 1) * 32, (kt + 1) & 1);  // issue only

    v16h bfrag[2];
#pragma unroll
    for (int nt = 0; nt < 2; ++nt)
      bfrag[nt] = load_frag_f16(&Ws[cur][wn * 32 + nt * 16 + lr][0], hgrp);
#pragma unroll
    for (int mt = 0; mt < 4; ++mt) {
      v16h afrag = load_frag_f16(&As[cur][wm * 64 + mt * 16 + lr][0], hgrp);
#pragma unroll
      for (int nt = 0; nt < 2; ++nt)
        acc[mt][nt] = __builtin_amdgcn_wmma_f32_16x16x32_f16(
            false, afrag, false, bfrag[nt], (short)0, acc[mt][nt], false, false);
    }

    if (kt + 1 < NKT) {
      asm volatile("s_wait_asynccnt 0x0" ::: "memory");
      __syncthreads();
    }
  }

  // epilogue: C layout lane 0-15 rows 0..7 (VGPR=r), lanes 16-31 rows 8..15
  const int rsel = (lane < 16) ? 0 : 8;
  const int ncol = lane & 15;
#pragma unroll
  for (int mt = 0; mt < 4; ++mt)
#pragma unroll
    for (int nt = 0; nt < 2; ++nt) {
      int row0 = mBase + wm * 64 + mt * 16 + rsel;
      int col  = nBase + wn * 32 + nt * 16 + ncol;
#pragma unroll
      for (int r = 0; r < 8; ++r)
        C[(size_t)(row0 + r) * NPROJ + col] = (_Float16)acc[mt][nt][r];
    }
}

// ---------------- persistent recurrent kernel ----------------
// Grid = 32 WGs: WGs [0..15] = forward dir, [16..31] = backward dir.
// Each WG owns a 32-wide hidden slice; per step computes the [64 x 128]
// gate slice (i/f/g/o columns for its hidden units) with WMMA from the
// ping-pong f16 h buffer (L2) and the L2-resident f16 W_hh, keeps cell
// state in LDS, then arrives at a per-step agent-scope counter.
__global__ __launch_bounds__(256) void lstm_rec(
    const _Float16* __restrict__ xproj,  // [T*B, 4096]
    const _Float16* __restrict__ whh,    // [2, 2048, 512] f16
    const float* __restrict__ bih,       // [2, 2048]
    const float* __restrict__ bhh,       // [2, 2048]
    const int*  __restrict__ mask,       // [B, T]
    _Float16* __restrict__ hbuf,         // [2 phase][2 dir][64][512] f16
    _Float16* __restrict__ out16,        // [T, B, 1024] f16
    float* __restrict__ out32,           // [T, B, 1024] f32 or nullptr
    unsigned* __restrict__ syncc) {      // [2, T]
  __shared__ float gates[BB][128];
  __shared__ float cstate[BB][HSLICE];

  const int tid  = threadIdx.x;
  const int wave = tid >> 5, lane = tid & 31;
  const int lr = lane & 15, hgrp = lane >> 4;
  const int dir = blockIdx.x >> 4;
  const int wg  = blockIdx.x & 15;
  const int hs  = wg * HSLICE;

  for (int i = tid; i < BB * HSLICE; i += 256) (&cstate[0][0])[i] = 0.0f;
  __syncthreads();

  // This wave's N-tile = wave (0..7): gate group gnum, 16 W_hh rows.
  const int gnum  = wave >> 1;
  const int wbase = gnum * 512 + hs + (wave & 1) * 16;
  const _Float16* wrow = whh + (size_t)(dir * GATES + wbase + lr) * HH;

  unsigned* myc = syncc + dir * TT;
  const v8f vzero = {0.f,0.f,0.f,0.f,0.f,0.f,0.f,0.f};

  for (int step = 0; step < TT; ++step) {
    if (step > 0) {
      if (tid == 0) {
        while (__hip_atomic_load(&myc[step - 1], __ATOMIC_ACQUIRE,
                                 __HIP_MEMORY_SCOPE_AGENT) < NWG_PER_DIR)
          __builtin_amdgcn_s_sleep(1);
      }
      __syncthreads();
    }
    const int t  = dir ? (TT - 1 - step) : step;
    const int rb = step & 1;
    const _Float16* hin = hbuf + (size_t)((rb * 2 + dir) * BB) * HH;

    // Prefetch this step's xproj gate slice (one 64B line per thread:
    // thread -> (batch=tid&63, gate=tid>>6)) so it is L2/L0-hot by the
    // time the elementwise phase runs after the WMMAs.
    {
      int pb = tid & 63, pg = tid >> 6;
      __builtin_prefetch(
          xproj + (size_t)(t * BB + pb) * NPROJ + dir * GATES + pg * 512 + hs,
          0, 1);
    }

    // [64 x 128] = h[64 x 512] * Whh_slice^T, K-loop of 16 WMMAs per wave
    v8f acc[4] = {vzero, vzero, vzero, vzero};
#pragma unroll 4
    for (int kk = 0; kk < HH; kk += 32) {
      v16h bfrag = load_frag_f16(wrow + kk, hgrp);
#pragma unroll
      for (int mt = 0; mt < 4; ++mt) {
        v16h afrag = load_frag_f16(hin + (size_t)(mt * 16 + lr) * HH + kk, hgrp);
        acc[mt] = __builtin_amdgcn_wmma_f32_16x16x32_f16(
            false, afrag, false, bfrag, (short)0, acc[mt], false, false);
      }
    }

    // stage gate pre-activations to LDS
    const int rsel = (lane < 16) ? 0 : 8;
    const int ncol = wave * 16 + (lane & 15);
#pragma unroll
    for (int mt = 0; mt < 4; ++mt)
#pragma unroll
      for (int r = 0; r < 8; ++r)
        gates[mt * 16 + rsel + r][ncol] = acc[mt][r];
    __syncthreads();

    // elementwise LSTM cell: 64 batch x 32 hidden = 2048 items
#pragma unroll
    for (int it = 0; it < 8; ++it) {
      int idx = tid + it * 256;
      int b = idx >> 5;
      int j = idx & 31;
      const _Float16* xp = xproj + (size_t)(t * BB + b) * NPROJ + dir * GATES;
      int gc = dir * GATES + hs + j;
      float bI = bih[gc]        + bhh[gc];
      float bF = bih[gc + 512]  + bhh[gc + 512];
      float bG = bih[gc + 1024] + bhh[gc + 1024];
      float bO = bih[gc + 1536] + bhh[gc + 1536];
      float gi = gates[b][j]      + (float)xp[hs + j]        + bI;
      float gf = gates[b][32 + j] + (float)xp[512 + hs + j]  + bF;
      float gg = gates[b][64 + j] + (float)xp[1024 + hs + j] + bG;
      float go = gates[b][96 + j] + (float)xp[1536 + hs + j] + bO;
      float c  = cstate[b][j];
      float cn = sigmoidf_(gf) * c + sigmoidf_(gi) * tanhf(gg);
      float hn = sigmoidf_(go) * tanhf(cn);
      float m  = (float)mask[b * TT + t];
      hn *= m; cn *= m;
      cstate[b][j] = cn;
      int wbp = (step + 1) & 1;
      hbuf[(size_t)((wbp * 2 + dir) * BB + b) * HH + hs + j] = (_Float16)hn;
      size_t oidx = (size_t)(t * BB + b) * 1024 + dir * HH + hs + j;
      out16[oidx] = (_Float16)hn;
      if (out32) out32[oidx] = hn;
    }
    __threadfence();
    __syncthreads();
    if (tid == 0)
      __hip_atomic_fetch_add(&myc[step], 1u, __ATOMIC_RELEASE,
                             __HIP_MEMORY_SCOPE_AGENT);
  }
}

// ---------------- host ----------------
extern "C" void kernel_launch(void* const* d_in, const int* in_sizes, int n_in,
                              void* d_out, int out_size, void* d_ws,
                              size_t ws_size, hipStream_t stream) {
  (void)in_sizes; (void)n_in; (void)out_size; (void)ws_size;
  const float* x    = (const float*)d_in[0];
  const int*   mask = (const int*)d_in[1];
  const float* w_ih = (const float*)d_in[2];
  const float* w_hh = (const float*)d_in[3];
  const float* b_ih = (const float*)d_in[4];
  const float* b_hh = (const float*)d_in[5];
  float* out = (float*)d_out;

  char* ws = (char*)d_ws;
  size_t off = 0;
  auto alloc = [&](size_t bytes) -> void* {
    void* p = ws + off;
    off = (off + bytes + 255) & ~(size_t)255;
    return p;
  };
  _Float16* xh16    = (_Float16*)alloc((size_t)TT * BB * INK * 2);     // 33.5 MB
  _Float16* xproj16 = (_Float16*)alloc((size_t)TT * BB * NPROJ * 2);   // 134 MB
  _Float16* wih16   = (_Float16*)alloc((size_t)NPROJ * INK * 2);       // 8.4 MB
  _Float16* whh16   = (_Float16*)alloc((size_t)2 * GATES * HH * 2);    // 4.2 MB
  _Float16* hbuf    = (_Float16*)alloc((size_t)2 * 2 * BB * HH * 2);   // 256 KB
  unsigned* syncc   = (unsigned*)alloc((size_t)2 * TT * 4);            // 2 KB

  // layer 0 input -> f16
  cvt_f32_f16<<<2048, 256, 0, stream>>>(x, xh16, TT * BB * INK);

  for (int l = 0; l < 2; ++l) {
    cvt_f32_f16<<<2048, 256, 0, stream>>>(w_ih + (size_t)l * NPROJ * INK,
                                          wih16, NPROJ * INK);
    cvt_f32_f16<<<2048, 256, 0, stream>>>(w_hh + (size_t)l * 2 * GATES * HH,
                                          whh16, 2 * GATES * HH);
    dim3 gg(TT * BB / 128, NPROJ / 128);
    xproj_gemm_f16<<<gg, 256, 0, stream>>>(xh16, wih16, xproj16);
    // zero ping-pong h (phase 0 read at step 0) + per-step sync counters
    // (hbuf and syncc are contiguous in ws)
    zero_u32<<<260, 256, 0, stream>>>((unsigned*)hbuf,
                                      (2 * 2 * BB * HH * 2 + 2 * TT * 4) / 4);
    lstm_rec<<<32, 256, 0, stream>>>(xproj16, whh16, b_ih + l * 2 * GATES,
                                     b_hh + l * 2 * GATES, mask, hbuf,
                                     /*out16=*/xh16,
                                     /*out32=*/(l == 1) ? out : nullptr, syncc);
  }
}
